// SelfAttentionBlock_25726854103545
// MI455X (gfx1250) — matrix operand order repaired and checked by reading, not compiled, not run on hardware
//
#include <hip/hip_runtime.h>
#include <hip/hip_bf16.h>
#include <math.h>

// ---------------------------------------------------------------------------
// Graph self-attention (SelfAttentionBlock) for MI455X / gfx1250, wave32.
//   qkv   = x @ Wqkv + b                      (f16 WMMA, f32 accum)
//   rpe   = edge_attr @ W{k,q}_rpe + b        (fused f16 WMMA in edge kernel)
//   compat= <q[s]*scale + q_rpe, k[t] + k_rpe> per head
//   attn  = segment_softmax(compat, by s)     (atomic max / atomic add)
//   out   = segment_sum(v[t] * attn, by s)    (atomic f32 add)
// Bandwidth-bound (~2.9 GB total => ~125us at 23.3 TB/s); WMMA used in-register
// on f32 data so HBM traffic stays fp32-exact while matrix math rides the XDL.
// ---------------------------------------------------------------------------

typedef __attribute__((ext_vector_type(16))) _Float16 v16h;
typedef __attribute__((ext_vector_type(8)))  float    v8f;

#define DIMX  256
#define NHEAD 16
#define DHEAD 8
#define DH    128
#define QKVC  512      // 2*DH + DIM
#define RPE   18
#define QK_SCALE 0.35355339059327373f   // 1/sqrt(8)

// float atomic max via sign-split integer atomics (lowers to
// global_atomic_max_i32 / global_atomic_min_u32 — single L2 atomic, no CAS loop)
__device__ __forceinline__ void atomicMaxF(float* addr, float val) {
  if (val >= 0.0f) atomicMax((int*)addr, __float_as_int(val));
  else             atomicMin((unsigned int*)addr, __float_as_uint(val));
}

__global__ void fill_neginf(float* __restrict__ p, long long n) {
  long long i = (long long)blockIdx.x * blockDim.x + threadIdx.x;
  if (i < n) p[i] = -INFINITY;
}

// ---------------------------------------------------------------------------
// Kernel 2: qkv = x @ Wqkv + bqkv  (N x 256) @ (256 x 512)
// 256 threads = 8 waves; wave owns one 16x16 output tile; K loop = 8 WMMAs,
// fully unrolled so loads/cvt of step i+1 co-issue with WMMA i.
// ---------------------------------------------------------------------------
__global__ __launch_bounds__(256) void qkv_gemm(
    const float* __restrict__ x, const float* __restrict__ W,
    const float* __restrict__ b, float* __restrict__ qkv, int N) {
  const int wave = threadIdx.x >> 5;
  const int lane = threadIdx.x & 31;
  const int n16  = lane & 15;
  const int hi   = lane >> 4;

  const int rowBase = blockIdx.x * 16;
  const int colBase = (blockIdx.y * 8 + wave) * 16;
  const int col     = colBase + n16;

  int rowL = rowBase + n16;                // A-load row (clamped: EXEC must stay full)
  if (rowL > N - 1) rowL = N - 1;
  const float* xr = x + (size_t)rowL * DIMX;

  const float bias = b[col];
  v8f c;
  #pragma unroll
  for (int r = 0; r < 8; ++r) c[r] = bias; // D(m,n) bias depends only on col

  #pragma unroll
  for (int kk = 0; kk < DIMX; kk += 32) {
    // A 16x32 f16: lane<16 -> K {0..7,16..23}; lane>=16 -> K {8..15,24..31}
    const int ka = kk + hi * 8;
    const float4 a0 = *(const float4*)(xr + ka);
    const float4 a1 = *(const float4*)(xr + ka + 4);
    const float4 a2 = *(const float4*)(xr + ka + 16);
    const float4 a3 = *(const float4*)(xr + ka + 20);
    v16h a, bb;
    a[0]  = (_Float16)a0.x; a[1]  = (_Float16)a0.y; a[2]  = (_Float16)a0.z; a[3]  = (_Float16)a0.w;
    a[4]  = (_Float16)a1.x; a[5]  = (_Float16)a1.y; a[6]  = (_Float16)a1.z; a[7]  = (_Float16)a1.w;
    a[8]  = (_Float16)a2.x; a[9]  = (_Float16)a2.y; a[10] = (_Float16)a2.z; a[11] = (_Float16)a2.w;
    a[12] = (_Float16)a3.x; a[13] = (_Float16)a3.y; a[14] = (_Float16)a3.z; a[15] = (_Float16)a3.w;
    // B 32x16 f16: lanes 0-15 hold K {0..7,16..23} of their column, lanes 16-31 K {8..15,24..31}
    const int kb = kk + hi * 8;
    #pragma unroll
    for (int j = 0; j < 16; ++j)
      bb[j] = (_Float16)W[(size_t)(kb + (j & 7) + ((j >> 3) << 4)) * QKVC + col];

    c = __builtin_amdgcn_wmma_f32_16x16x32_f16(false, a, false, bb,
                                               (short)0, c, false, false);
  }
  // C/D layout: VGPR r, lane -> (M = r + 8*hi, N = lane&15)
  if (rowBase + 16 <= N) {                 // uniform branch: full tile (common case)
    #pragma unroll
    for (int r = 0; r < 8; ++r)
      qkv[(size_t)(rowBase + r + hi * 8) * QKVC + col] = c[r];
  } else {                                 // ragged tail (never taken for N%16==0)
    #pragma unroll
    for (int r = 0; r < 8; ++r) {
      const int row = rowBase + r + hi * 8;
      if (row < N) qkv[(size_t)row * QKVC + col] = c[r];
    }
  }
}

// ---------------------------------------------------------------------------
// Kernel 3: fused RPE GEMMs + gather + compat + atomic running max.
// 64 threads = 2 waves; each wave owns 16 edges. RPE K=18 zero-padded to 32.
// Fusion avoids materializing k_rpe/q_rpe (would be +1.6 GB of HBM traffic).
// ---------------------------------------------------------------------------
__global__ __launch_bounds__(64) void edge_compat(
    const float* __restrict__ attr, const int* __restrict__ eidx,
    const float* __restrict__ Wk, const float* __restrict__ bk,
    const float* __restrict__ Wq, const float* __restrict__ bq,
    const float* __restrict__ qkv, float* __restrict__ compat,
    float* __restrict__ mbuf, long long E) {
  __shared__ float s_k[2][16][DH];
  __shared__ float s_q[2][16][DH];

  const int wave = threadIdx.x >> 5;
  const int lane = threadIdx.x & 31;
  const int n16  = lane & 15;
  const int hi   = lane >> 4;

  const long long e0 = ((long long)blockIdx.x * 2 + wave) * 16;

  // ---- A tile: 16 edges x 32 (edge_attr, zero-padded past K=17) ----
  long long eA = e0 + n16;
  if (eA > E - 1) eA = E - 1;              // clamp: keep EXEC all-ones for WMMA
  const float* ar = attr + eA * RPE;
  v16h a;
  #pragma unroll
  for (int j = 0; j < 8; ++j) {
    const int k1 = hi * 8 + j;             // 0..15
    const int k2 = hi * 8 + 16 + j;        // 16..31
    a[j]     = (_Float16)(k1 < RPE ? ar[k1] : 0.0f);
    a[8 + j] = (_Float16)(k2 < RPE ? ar[k2] : 0.0f);
  }

  // ---- 8 column tiles x (k_rpe, q_rpe) = 16 WMMAs per 16 edges ----
  #pragma unroll
  for (int ct = 0; ct < 8; ++ct) {
    const int col = ct * 16 + n16;
    const int kb  = hi * 8;
    v16h bbk, bbq;
    #pragma unroll
    for (int j = 0; j < 16; ++j) {
      const int k = kb + (j & 7) + ((j >> 3) << 4);
      bbk[j] = (_Float16)(k < RPE ? Wk[k * DH + col] : 0.0f);
      bbq[j] = (_Float16)(k < RPE ? Wq[k * DH + col] : 0.0f);
    }
    const float bkv = bk[col], bqv = bq[col];
    v8f ck, cq;
    #pragma unroll
    for (int r = 0; r < 8; ++r) { ck[r] = bkv; cq[r] = bqv; }
    ck = __builtin_amdgcn_wmma_f32_16x16x32_f16(false, a, false, bbk,
                                                (short)0, ck, false, false);
    cq = __builtin_amdgcn_wmma_f32_16x16x32_f16(false, a, false, bbq,
                                                (short)0, cq, false, false);
    #pragma unroll
    for (int r = 0; r < 8; ++r) {
      s_k[wave][r + hi * 8][col] = ck[r];
      s_q[wave][r + hi * 8][col] = cq[r];
    }
  }
  __syncthreads();

  // ---- compat: 16 edges x 16 heads = 256 pairs; 8 per lane ----
  const int m = lane >> 1;                 // edge-in-tile
  const long long e = e0 + m;
  const bool valid = (e < E);
  const long long ec = valid ? e : (E - 1);
  const int s = eidx[ec];
  const int t = eidx[E + ec];
  const float* qrow = qkv + (size_t)s * QKVC;        // q at cols [0,128)
  const float* krow = qkv + (size_t)t * QKVC + DH;   // k at cols [128,256)
  const int hbase = (lane & 1) * 8;

  // vector-gather this lane's 8 heads worth of q and k (2x b128 per head pair)
  #pragma unroll
  for (int i = 0; i < 8; ++i) {
    const int h = hbase + i;
    const float4 q0 = *(const float4*)(qrow + h * DHEAD);
    const float4 q1 = *(const float4*)(qrow + h * DHEAD + 4);
    const float4 k0 = *(const float4*)(krow + h * DHEAD);
    const float4 k1 = *(const float4*)(krow + h * DHEAD + 4);
    const float* sq = &s_q[wave][m][h * DHEAD];
    const float* sk = &s_k[wave][m][h * DHEAD];
    float acc = 0.0f;
    acc += (q0.x * QK_SCALE + sq[0]) * (k0.x + sk[0]);
    acc += (q0.y * QK_SCALE + sq[1]) * (k0.y + sk[1]);
    acc += (q0.z * QK_SCALE + sq[2]) * (k0.z + sk[2]);
    acc += (q0.w * QK_SCALE + sq[3]) * (k0.w + sk[3]);
    acc += (q1.x * QK_SCALE + sq[4]) * (k1.x + sk[4]);
    acc += (q1.y * QK_SCALE + sq[5]) * (k1.y + sk[5]);
    acc += (q1.z * QK_SCALE + sq[6]) * (k1.z + sk[6]);
    acc += (q1.w * QK_SCALE + sq[7]) * (k1.w + sk[7]);
    if (valid) {
      compat[e * 16 + h] = acc;
      atomicMaxF(&mbuf[(size_t)s * 16 + h], acc);
    }
  }
}

// ---------------------------------------------------------------------------
// Kernel 4: one thread per edge: ex = exp(compat - m[s]); den[s,:] += ex.
// compat & mbuf rows are contiguous -> b128 loads; compat overwritten in place.
// ---------------------------------------------------------------------------
__global__ void exp_den(const int* __restrict__ eidx, float* __restrict__ compat,
                        const float* __restrict__ mbuf, float* __restrict__ den,
                        long long E) {
  const long long e = (long long)blockIdx.x * blockDim.x + threadIdx.x;
  if (e >= E) return;
  const int s = eidx[e];
  float4* crow = (float4*)(compat + e * 16);
  const float4* mrow = (const float4*)(mbuf + (size_t)s * 16);
  float* drow = den + (size_t)s * 16;
  #pragma unroll
  for (int q = 0; q < 4; ++q) {
    const float4 cc = crow[q];
    const float4 mm = mrow[q];
    const float e0 = __expf(cc.x - mm.x);
    const float e1 = __expf(cc.y - mm.y);
    const float e2 = __expf(cc.z - mm.z);
    const float e3 = __expf(cc.w - mm.w);
    crow[q] = make_float4(e0, e1, e2, e3);
    atomicAdd(&drow[q * 4 + 0], e0);
    atomicAdd(&drow[q * 4 + 1], e1);
    atomicAdd(&drow[q * 4 + 2], e2);
    atomicAdd(&drow[q * 4 + 3], e3);
  }
}

// ---------------------------------------------------------------------------
// Kernel 5: out[s,:] += v[t,:] * ex/(den[s]+eps). 32 lanes per edge,
// one 8-float chunk (within a single head) per lane; v gathered as 2x b128.
// ---------------------------------------------------------------------------
__global__ __launch_bounds__(256) void aggregate(
    const int* __restrict__ eidx, const float* __restrict__ qkv,
    const float* __restrict__ ex, const float* __restrict__ den,
    float* __restrict__ out, long long E) {
  const long long e = (long long)blockIdx.x * 8 + (threadIdx.x >> 5);
  if (e >= E) return;
  const int lane = threadIdx.x & 31;
  const int c0 = lane * 8;                 // column chunk [c0, c0+8) in [0,256)
  const int h  = lane >> 1;                // c0/16
  const int s = eidx[e];
  const int t = eidx[E + e];
  const float w = ex[e * 16 + h] / (den[(size_t)s * 16 + h] + 1e-16f);
  const float* vr = qkv + (size_t)t * QKVC + 256 + c0;  // v at cols [256,512)
  const float4 v0 = *(const float4*)(vr);
  const float4 v1 = *(const float4*)(vr + 4);
  float* orow = out + (size_t)s * 256 + c0;
  atomicAdd(&orow[0], v0.x * w);
  atomicAdd(&orow[1], v0.y * w);
  atomicAdd(&orow[2], v0.z * w);
  atomicAdd(&orow[3], v0.w * w);
  atomicAdd(&orow[4], v1.x * w);
  atomicAdd(&orow[5], v1.y * w);
  atomicAdd(&orow[6], v1.z * w);
  atomicAdd(&orow[7], v1.w * w);
}

// ---------------------------------------------------------------------------
extern "C" void kernel_launch(void* const* d_in, const int* in_sizes, int n_in,
                              void* d_out, int out_size, void* d_ws, size_t ws_size,
                              hipStream_t stream) {
  const float* x    = (const float*)d_in[0];
  const int*   eidx = (const int*)  d_in[1];
  const float* attr = (const float*)d_in[2];
  const float* Wqkv = (const float*)d_in[3];
  const float* bqkv = (const float*)d_in[4];
  const float* Wk   = (const float*)d_in[5];
  const float* bk   = (const float*)d_in[6];
  const float* Wq   = (const float*)d_in[7];
  const float* bq   = (const float*)d_in[8];

  const int       N = in_sizes[0] / DIMX;     // 50000
  const long long E = in_sizes[1] / 2;        // 800000

  // workspace layout (≈160 MB)
  float* qkv    = (float*)d_ws;               // N x 512
  float* compat = qkv    + (size_t)N * QKVC;  // E x 16 (becomes ex in-place)
  float* mbuf   = compat + (size_t)E * 16;    // N x 16 running max
  float* den    = mbuf   + (size_t)N * 16;    // N x 16 softmax denominator

  hipMemsetAsync(d_out, 0, (size_t)out_size * sizeof(float), stream);
  hipMemsetAsync(den, 0, (size_t)N * 16 * sizeof(float), stream);
  const long long nm = (long long)N * 16;
  fill_neginf<<<(unsigned)((nm + 255) / 256), 256, 0, stream>>>(mbuf, nm);

  dim3 g1((unsigned)((N + 15) / 16), 4);      // 8 waves/block cover 8 col tiles
  qkv_gemm<<<g1, 256, 0, stream>>>(x, Wqkv, bqkv, qkv, N);

  edge_compat<<<(unsigned)((E + 31) / 32), 64, 0, stream>>>(
      attr, eidx, Wk, bk, Wq, bq, qkv, compat, mbuf, E);

  exp_den<<<(unsigned)((E + 255) / 256), 256, 0, stream>>>(eidx, compat, mbuf, den, E);

  aggregate<<<(unsigned)((E + 7) / 8), 256, 0, stream>>>(
      eidx, qkv, compat, den, (float*)d_out, E);
}
